// MultiHeadAttention_21586505630034
// MI455X (gfx1250) — matrix operand order repaired and checked by reading, not compiled, not run on hardware
//
#include <hip/hip_runtime.h>
#include <hip/hip_bf16.h>

#define DI __device__ __forceinline__

typedef __attribute__((ext_vector_type(16))) __bf16 v16bf;
typedef __attribute__((ext_vector_type(8)))  __bf16 v8bf;
typedef __attribute__((ext_vector_type(8)))  float  v8f;
typedef __attribute__((ext_vector_type(4)))  int    v4i;

constexpr int B = 2, S = 4096, D = 768, H = 12, HD = 64;
constexpr int M   = B * S;     // 8192 activation rows
constexpr int NT2 = D / 32;    // 24 col blocks (32 wide)
constexpr int MT2 = M / 32;    // 256 row blocks (32 tall)

// ---- workspace layout (bytes) ----
constexpr size_t OFF_XB = 0;
constexpr size_t SZ_XB  = (size_t)M * D * 2;          // x in bf16
constexpr size_t OFF_WT = OFF_XB + SZ_XB;             // 4 transposed bf16 weight mats
constexpr size_t SZ_W1  = (size_t)D * D * 2;
constexpr size_t OFF_Q  = OFF_WT + 4 * SZ_W1;         // Q  [b][h][s][64] bf16
constexpr size_t OFF_K  = OFF_Q + SZ_XB;              // K  [b][h][s][64] bf16
constexpr size_t OFF_VT = OFF_K + SZ_XB;              // Vt [b][h][64][s] bf16
constexpr size_t OFF_O  = OFF_VT + SZ_XB;             // O  [m][768] f32

#if __has_builtin(__builtin_amdgcn_global_load_async_to_lds_b128)
#define HAS_ASYNC_LDS 1
#else
#define HAS_ASYNC_LDS 0
#endif

typedef __attribute__((address_space(1))) v4i Gv4i;   // global int4
typedef __attribute__((address_space(3))) v4i Lv4i;   // LDS int4

// 16-byte global -> LDS copy; async DMA (ASYNCcnt) when available.
DI void async_cp16(void* lds_dst, const void* gsrc) {
#if HAS_ASYNC_LDS
  __builtin_amdgcn_global_load_async_to_lds_b128((Gv4i*)gsrc, (Lv4i*)lds_dst, 0, 0);
#else
  *(v8bf*)lds_dst = *(const v8bf*)gsrc;
#endif
}

DI void wait_async0() {
#if HAS_ASYNC_LDS
#if __has_builtin(__builtin_amdgcn_s_wait_asynccnt)
  __builtin_amdgcn_s_wait_asynccnt(0);
#else
  asm volatile("s_wait_asynccnt 0x0" ::: "memory");
#endif
#endif
}

DI v8f wmma_bf16(v16bf a, v16bf b, v8f c) {
  return __builtin_amdgcn_wmma_f32_16x16x32_bf16(false, a, false, b, (short)0, c,
                                                 false, false);
}

// A-fragment (16x32 bf16, row-major). Lane r = lane&15 owns row r.
// elements 0..15 = row[g*16 .. g*16+15]
DI v16bf load_a_frag(const __bf16* rowk, int g) {
  v8bf lo = *(const v8bf*)(rowk + g * 16);
  v8bf hi = *(const v8bf*)(rowk + g * 16 + 8);
  return __builtin_shufflevector(lo, hi, 0, 1, 2, 3, 4, 5, 6, 7,
                                 8, 9, 10, 11, 12, 13, 14, 15);
}

// B-fragment: 16 contiguous bf16 from the lane's column, as two 16B loads
// (safe for 16B-aligned padded LDS rows).
DI v16bf load_b_frag(const __bf16* p) {
  v8bf lo = *(const v8bf*)(p);
  v8bf hi = *(const v8bf*)(p + 8);
  return __builtin_shufflevector(lo, hi, 0, 1, 2, 3, 4, 5, 6, 7,
                                 8, 9, 10, 11, 12, 13, 14, 15);
}

// A-fragment with f32 -> bf16 conversion on load (for the O @ Wo GEMM).
DI v16bf cvt_a_frag(const float* rowk, int g) {
  v8f lo = *(const v8f*)(rowk + g * 16);
  v8f hi = *(const v8f*)(rowk + g * 16 + 8);
  v16bf r;
#pragma unroll
  for (int i = 0; i < 8; ++i) {
    r[i]     = (__bf16)lo[i];
    r[8 + i] = (__bf16)hi[i];
  }
  return r;
}

// ---------------- conversion kernels ----------------
__global__ void convert_x_kernel(const float* __restrict__ x,
                                 __bf16* __restrict__ xb, int nElem) {
  int i = blockIdx.x * blockDim.x + threadIdx.x;
  if (i < nElem) xb[i] = (__bf16)x[i];
}

__global__ void convert_wT_kernel(const float* __restrict__ W0,
                                  const float* __restrict__ W1,
                                  const float* __restrict__ W2,
                                  const float* __restrict__ W3,
                                  __bf16* __restrict__ WtAll) {
  const int which = blockIdx.y;
  const float* W = (which == 0) ? W0 : (which == 1) ? W1 : (which == 2) ? W2 : W3;
  __bf16* Wt = WtAll + (size_t)which * D * D;
  int i = blockIdx.x * blockDim.x + threadIdx.x;
  if (i < D * D) {
    int nn = i / D, kk = i % D;
    Wt[i] = (__bf16)W[(size_t)kk * D + nn];
  }
}

DI void store_qkv(int mode, __bf16* __restrict__ Qb, __bf16* __restrict__ Kb,
                  __bf16* __restrict__ Vtb, int mrow, int col, float v) {
  const int bb = mrow >> 12, ss = mrow & (S - 1);
  const int h = col >> 6, dd = col & 63;
  if (mode == 2) {  // V stored transposed: [b][h][64][s]
    Vtb[(((size_t)bb * H + h) * HD + dd) * S + ss] = (__bf16)v;
  } else {          // Q/K stored [b][h][s][64]
    __bf16* dst = (mode == 0) ? Qb : Kb;
    dst[(((size_t)bb * H + h) * S + ss) * HD + dd] = (__bf16)v;
  }
}

// ---------------- QKV projection: X @ W + b (32x32 block per wave) ----------
__global__ void __launch_bounds__(128)
qkv_gemm(const __bf16* __restrict__ Xb, const __bf16* __restrict__ WtAll,
         const float* __restrict__ bq, const float* __restrict__ bk,
         const float* __restrict__ bv, __bf16* __restrict__ Qb,
         __bf16* __restrict__ Kb, __bf16* __restrict__ Vtb) {
  const int lane = threadIdx.x & 31, wave = threadIdx.x >> 5;
  const int g = lane >> 4, n = lane & 15;
  const int mode = blockIdx.z;
  const int blk = blockIdx.x * 4 + wave;
  const int tN = blk % NT2, tM = blk / NT2;

  const __bf16* Wt   = WtAll + (size_t)mode * D * D;
  const float*  bias = (mode == 0) ? bq : (mode == 1) ? bk : bv;
  const __bf16* arow0 = Xb + (size_t)(tM * 32 + n) * D;
  const __bf16* arow1 = arow0 + (size_t)16 * D;
  const __bf16* bcol0 = Wt + (size_t)(tN * 32 + n) * D;
  const __bf16* bcol1 = bcol0 + (size_t)16 * D;

  v8f a00 = {}, a01 = {}, a10 = {}, a11 = {};
#pragma unroll 2
  for (int kt = 0; kt < D; kt += 32) {
    v16bf af0 = load_a_frag(arow0 + kt, g);
    v16bf af1 = load_a_frag(arow1 + kt, g);
    v16bf bf0 = load_b_frag(bcol0 + kt + g * 16);
    v16bf bf1 = load_b_frag(bcol1 + kt + g * 16);
    a00 = wmma_bf16(af0, bf0, a00);
    a01 = wmma_bf16(af0, bf1, a01);
    a10 = wmma_bf16(af1, bf0, a10);
    a11 = wmma_bf16(af1, bf1, a11);
  }

  const int col0 = tN * 32 + n, col1 = col0 + 16;
  const float bv0 = bias[col0], bv1 = bias[col1];
#pragma unroll
  for (int p = 0; p < 8; ++p) {
    const int row0 = tM * 32 + p + 8 * g, row1 = row0 + 16;
    store_qkv(mode, Qb, Kb, Vtb, row0, col0, a00[p] + bv0);
    store_qkv(mode, Qb, Kb, Vtb, row0, col1, a01[p] + bv1);
    store_qkv(mode, Qb, Kb, Vtb, row1, col0, a10[p] + bv0);
    store_qkv(mode, Qb, Kb, Vtb, row1, col1, a11[p] + bv1);
  }
}

// ---------------- flash attention with async double-buffered K/V tiles -----
__global__ void __launch_bounds__(128)
attn_kernel(const __bf16* __restrict__ Q, const __bf16* __restrict__ K,
            const __bf16* __restrict__ Vt, float* __restrict__ O) {
  // padded rows (72/40 bf16) to rotate LDS bank groups
  __shared__ __align__(16) __bf16 Kt[2][32][72];
  __shared__ __align__(16) __bf16 Vtile[2][64][40];
  __shared__ __align__(32) __bf16 Pld[4][16][32];

  const int tid = threadIdx.x;
  const int lane = tid & 31, wave = tid >> 5;
  const int g = lane >> 4, n = lane & 15;
  const int b = blockIdx.z, h = blockIdx.y;
  const int qbase = (blockIdx.x * 4 + wave) * 16;

  const __bf16* Qh = Q + (size_t)(b * H + h) * S * HD;
  const __bf16* Kh = K + (size_t)(b * H + h) * S * HD;
  const __bf16* Vh = Vt + (size_t)(b * H + h) * HD * S;

  // cooperative stage of one 32-wide j-chunk of K (32x64) and V (64x32)
  auto stage = [&](int buf, int j0) {
#pragma unroll
    for (int rep = 0; rep < 2; ++rep) {
      int idx = tid + rep * 128;          // 256 x 16B for K
      int row = idx >> 3, c = idx & 7;
      async_cp16(&Kt[buf][row][c * 8], Kh + (size_t)(j0 + row) * HD + c * 8);
    }
#pragma unroll
    for (int rep = 0; rep < 2; ++rep) {
      int idx = tid + rep * 128;          // 256 x 16B for V
      int row = idx >> 2, c = idx & 3;
      async_cp16(&Vtile[buf][row][c * 8], Vh + (size_t)row * S + j0 + c * 8);
    }
  };

  // Q fragments persist across the whole j loop
  v16bf qf0 = load_a_frag(Qh + (size_t)(qbase + n) * HD, g);
  v16bf qf1 = load_a_frag(Qh + (size_t)(qbase + n) * HD + 32, g);

  v8f acc0 = {}, acc1 = {}, acc2 = {}, acc3 = {};
  float m[8], l[8];
#pragma unroll
  for (int p = 0; p < 8; ++p) { m[p] = -3.0e38f; l[p] = 0.0f; }

  stage(0, 0);
  for (int j0 = 0; j0 < S; j0 += 32) {
    const int cur = (j0 >> 5) & 1;
    wait_async0();          // drain DMA for buffer `cur`
    __syncthreads();        // all waves see staged tile; prior reads of cur^1 done
    if (j0 + 32 < S) stage(cur ^ 1, j0 + 32);  // overlap next DMA with compute

    // scores: two 16x16 tiles, K-dim 64 = 2 WMMA steps each
    v8f s0 = {}, s1 = {};
    {
      v16bf kb0 = load_b_frag(&Kt[cur][n][g * 16]);
      v16bf kb1 = load_b_frag(&Kt[cur][n][32 + g * 16]);
      s0 = wmma_bf16(qf0, kb0, s0);
      s0 = wmma_bf16(qf1, kb1, s0);
      v16bf kb2 = load_b_frag(&Kt[cur][16 + n][g * 16]);
      v16bf kb3 = load_b_frag(&Kt[cur][16 + n][32 + g * 16]);
      s1 = wmma_bf16(qf0, kb2, s1);
      s1 = wmma_bf16(qf1, kb3, s1);
    }

    // online softmax; row p+8g lives in the 16 lanes of half g in VGPR p
#pragma unroll
    for (int p = 0; p < 8; ++p) {
      float a = s0[p] * 8.0f;  // faithful to reference: * sqrt(64)
      float c = s1[p] * 8.0f;
      float mx = fmaxf(a, c);
      mx = fmaxf(mx, __shfl_xor(mx, 1, 32));
      mx = fmaxf(mx, __shfl_xor(mx, 2, 32));
      mx = fmaxf(mx, __shfl_xor(mx, 4, 32));
      mx = fmaxf(mx, __shfl_xor(mx, 8, 32));
      float mn = fmaxf(m[p], mx);
      float ea = __expf(a - mn), ec = __expf(c - mn);
      float rs = ea + ec;
      rs += __shfl_xor(rs, 1, 32);
      rs += __shfl_xor(rs, 2, 32);
      rs += __shfl_xor(rs, 4, 32);
      rs += __shfl_xor(rs, 8, 32);
      float corr = __expf(m[p] - mn);
      l[p] = l[p] * corr + rs;
      m[p] = mn;
      acc0[p] *= corr; acc1[p] *= corr; acc2[p] *= corr; acc3[p] *= corr;
      Pld[wave][p + 8 * g][n]      = (__bf16)ea;
      Pld[wave][p + 8 * g][16 + n] = (__bf16)ec;
    }
    __builtin_amdgcn_wave_barrier();  // wave-internal C->A relayout of P

    v16bf pf  = load_a_frag(&Pld[wave][n][0], g);
    v16bf vb0 = load_b_frag(&Vtile[cur][0 + n][g * 16]);
    v16bf vb1 = load_b_frag(&Vtile[cur][16 + n][g * 16]);
    v16bf vb2 = load_b_frag(&Vtile[cur][32 + n][g * 16]);
    v16bf vb3 = load_b_frag(&Vtile[cur][48 + n][g * 16]);
    acc0 = wmma_bf16(pf, vb0, acc0);
    acc1 = wmma_bf16(pf, vb1, acc1);
    acc2 = wmma_bf16(pf, vb2, acc2);
    acc3 = wmma_bf16(pf, vb3, acc3);
    __builtin_amdgcn_wave_barrier();
  }

#pragma unroll
  for (int p = 0; p < 8; ++p) {
    float inv = 1.0f / l[p];
    int row = qbase + p + 8 * g;
    float* orow = O + ((size_t)b * S + row) * D + h * HD + n;
    orow[0]  = acc0[p] * inv;
    orow[16] = acc1[p] * inv;
    orow[32] = acc2[p] * inv;
    orow[48] = acc3[p] * inv;
  }
}

// ---------------- output projection: O @ Wo + bo (32x32 block per wave) -----
__global__ void __launch_bounds__(128)
out_gemm(const float* __restrict__ Of, const __bf16* __restrict__ Wot,
         const float* __restrict__ bo, float* __restrict__ out) {
  const int lane = threadIdx.x & 31, wave = threadIdx.x >> 5;
  const int g = lane >> 4, n = lane & 15;
  const int blk = blockIdx.x * 4 + wave;
  const int tN = blk % NT2, tM = blk / NT2;

  const float*  arow0 = Of + (size_t)(tM * 32 + n) * D;
  const float*  arow1 = arow0 + (size_t)16 * D;
  const __bf16* bcol0 = Wot + (size_t)(tN * 32 + n) * D;
  const __bf16* bcol1 = bcol0 + (size_t)16 * D;

  v8f a00 = {}, a01 = {}, a10 = {}, a11 = {};
#pragma unroll 2
  for (int kt = 0; kt < D; kt += 32) {
    v16bf af0 = cvt_a_frag(arow0 + kt, g);
    v16bf af1 = cvt_a_frag(arow1 + kt, g);
    v16bf bf0 = load_b_frag(bcol0 + kt + g * 16);
    v16bf bf1 = load_b_frag(bcol1 + kt + g * 16);
    a00 = wmma_bf16(af0, bf0, a00);
    a01 = wmma_bf16(af0, bf1, a01);
    a10 = wmma_bf16(af1, bf0, a10);
    a11 = wmma_bf16(af1, bf1, a11);
  }

  const int col0 = tN * 32 + n, col1 = col0 + 16;
  const float bv0 = bo[col0], bv1 = bo[col1];
#pragma unroll
  for (int p = 0; p < 8; ++p) {
    const int row0 = tM * 32 + p + 8 * g, row1 = row0 + 16;
    out[(size_t)row0 * D + col0] = a00[p] + bv0;
    out[(size_t)row0 * D + col1] = a01[p] + bv1;
    out[(size_t)row1 * D + col0] = a10[p] + bv0;
    out[(size_t)row1 * D + col1] = a11[p] + bv1;
  }
}

// ---------------- host launcher ----------------
extern "C" void kernel_launch(void* const* d_in, const int* in_sizes, int n_in,
                              void* d_out, int out_size, void* d_ws, size_t ws_size,
                              hipStream_t stream) {
  (void)in_sizes; (void)n_in; (void)out_size; (void)ws_size;
  const float* x  = (const float*)d_in[0];
  const float* Wq = (const float*)d_in[1];
  const float* bq = (const float*)d_in[2];
  const float* Wk = (const float*)d_in[3];
  const float* bk = (const float*)d_in[4];
  const float* Wv = (const float*)d_in[5];
  const float* bv = (const float*)d_in[6];
  const float* Wo = (const float*)d_in[7];
  const float* bo = (const float*)d_in[8];

  char*   ws  = (char*)d_ws;
  __bf16* Xb  = (__bf16*)(ws + OFF_XB);
  __bf16* Wt  = (__bf16*)(ws + OFF_WT);
  __bf16* Qb  = (__bf16*)(ws + OFF_Q);
  __bf16* Kb  = (__bf16*)(ws + OFF_K);
  __bf16* Vtb = (__bf16*)(ws + OFF_VT);
  float*  Ob  = (float*)(ws + OFF_O);
  float*  out = (float*)d_out;

  convert_x_kernel<<<(M * D + 255) / 256, 256, 0, stream>>>(x, Xb, M * D);
  convert_wT_kernel<<<dim3((D * D + 255) / 256, 4), 256, 0, stream>>>(Wq, Wk, Wv, Wo, Wt);
  qkv_gemm<<<dim3(MT2 * NT2 / 4, 1, 3), 128, 0, stream>>>(Xb, Wt, bq, bk, bv, Qb, Kb, Vtb);
  attn_kernel<<<dim3(S / 64, H, B), 128, 0, stream>>>(Qb, Kb, Vtb, Ob);
  out_gemm<<<dim3(MT2 * NT2 / 4), 128, 0, stream>>>(Ob, Wt + 3 * (size_t)D * D, bo, out);
}
